// GPTAttention_37271726195318
// MI455X (gfx1250) — matrix operand order repaired and checked by reading, not compiled, not run on hardware
//
#include <hip/hip_runtime.h>
#include <hip/hip_fp16.h>
#include <math.h>

#define BB 4
#define NN 2048
#define DD 1024
#define HH 64

typedef _Float16 h8   __attribute__((ext_vector_type(8)));
typedef _Float16 v16h __attribute__((ext_vector_type(16)));
typedef float    v8f  __attribute__((ext_vector_type(8)));

// ---- WMMA fragment loaders (contiguous k order, wave32) -------------------
// A (16xK=32, MxK): lanes 0-15 hold row M=lane, K=0..15;
// lanes 16-31 hold row M=lane-16, K=16..31.
static __device__ __forceinline__ v16h load_a_frag(const _Float16* base, int ld,
                                                   int row0, int k0) {
  const int lane = threadIdx.x & 31;
  const int m    = lane & 15;
  const int hh   = lane >> 4;
  const _Float16* p = base + (size_t)(row0 + m) * ld + k0 + hh * 16;
  h8 lo = *(const h8*)(p);
  h8 hi = *(const h8*)(p + 8);
  return __builtin_shufflevector(lo, hi, 0,1,2,3,4,5,6,7,8,9,10,11,12,13,14,15);
}

// B (K=32 x 16, KxN): lanes 0-15 hold col N=lane, K=0..15; lanes 16-31 hold
// col N=lane-16, K=16..31.  Source must be column-K-contiguous: row n of
// `base` is column n of B (i.e. pass Wt[H][D], k[N][H], vT[H][N]).
static __device__ __forceinline__ v16h load_b_frag(const _Float16* base, int ld,
                                                   int n0, int k0) {
  const int lane = threadIdx.x & 31;
  const int n    = lane & 15;
  const int kb   = (lane >> 4) * 16;
  const _Float16* p = base + (size_t)(n0 + n) * ld + k0 + kb;
  h8 a = *(const h8*)(p);
  h8 b = *(const h8*)(p + 8);
  return __builtin_shufflevector(a, b, 0,1,2,3,4,5,6,7,8,9,10,11,12,13,14,15);
}

static __device__ __forceinline__ v8f wmma_f16(v16h a, v16h b, v8f c) {
  return __builtin_amdgcn_wmma_f32_16x16x32_f16(false, a, false, b,
                                                (short)0, c, false, false);
}

// ---- conversions ------------------------------------------------------------
__global__ void cvt_x_kernel(const float* __restrict__ x,
                             _Float16* __restrict__ xh, size_t n) {
  size_t i = ((size_t)blockIdx.x * blockDim.x + threadIdx.x) * 8;
  if (i >= n) return;
#pragma unroll
  for (int t = 0; t < 8; ++t) xh[i + t] = (_Float16)x[i + t];
}

__global__ void wt_kernel(const float* __restrict__ Wq,
                          const float* __restrict__ Wk,
                          const float* __restrict__ Wv,
                          _Float16* __restrict__ Wt) {
  int i = blockIdx.x * blockDim.x + threadIdx.x;     // over 3*H*D
  if (i >= 3 * HH * DD) return;
  int m = i / (HH * DD);
  int r = i % (HH * DD);
  int h = r / DD, d = r % DD;
  const float* W = (m == 0) ? Wq : ((m == 1) ? Wk : Wv);
  Wt[i] = (_Float16)W[(size_t)d * HH + h];           // Wt[m][h][d] = W[d][h]
}

// ---- QKV projection: [16 x 1024] @ [1024 x 64] per tile ---------------------
__global__ __launch_bounds__(128) void qkv_kernel(
    const _Float16* __restrict__ xh, const _Float16* __restrict__ Wt,
    const float* __restrict__ bq, const float* __restrict__ bk,
    const float* __restrict__ bv,
    _Float16* __restrict__ qh, _Float16* __restrict__ kh,
    _Float16* __restrict__ vt) {
  const int tile = blockIdx.x % (NN / 16);
  const int b    = blockIdx.x / (NN / 16);
  const int mat  = blockIdx.y;              // 0=q 1=k 2=v
  const int wv   = threadIdx.x >> 5;        // h-tile (4 waves)
  const int lane = threadIdx.x & 31;
  const int row0 = tile * 16;
  const _Float16* A0 = xh + (size_t)b * NN * DD;
  const _Float16* B0 = Wt + (size_t)mat * HH * DD + (size_t)(wv * 16) * DD;
  v8f acc = {};
  for (int k0 = 0; k0 < DD; k0 += 32)
    acc = wmma_f16(load_a_frag(A0, DD, row0, k0), load_b_frag(B0, DD, 0, k0), acc);
  const int h = wv * 16 + (lane & 15);
  const float* bias = (mat == 0) ? bq : ((mat == 1) ? bk : bv);
  const float bb = bias[h];
#pragma unroll
  for (int r = 0; r < 8; ++r) {             // D layout: M = r + 8*(lane>=16)
    int row   = row0 + r + 8 * (lane >> 4);
    float val = acc[r] + bb;
    if (mat == 0)      qh[((size_t)b * NN + row) * HH + h] = (_Float16)val;
    else if (mat == 1) kh[((size_t)b * NN + row) * HH + h] = (_Float16)val;
    else               vt[((size_t)b * HH + h) * NN + row] = (_Float16)val;   // vT
  }
}

// ---- Pass A: per-row softmax stats m[j], Z[j] over i >= j, s != 0 -----------
__global__ __launch_bounds__(32) void stats_kernel(
    const _Float16* __restrict__ qh, const _Float16* __restrict__ kh,
    float* __restrict__ mbuf, float* __restrict__ zbuf) {
  const int jt   = blockIdx.x % (NN / 16);
  const int b    = blockIdx.x / (NN / 16);
  const int lane = threadIdx.x & 31;
  const int row0 = jt * 16;
  const float scale = 0.125f;               // H^-0.5, H=64
  const _Float16* Q = qh + (size_t)b * NN * HH;
  const _Float16* K = kh + (size_t)b * NN * HH;
  float mx[8], sm[8];
#pragma unroll
  for (int r = 0; r < 8; ++r) { mx[r] = -INFINITY; sm[r] = 0.f; }
  for (int it = jt; it < NN / 16; ++it) {   // only i-tiles at/after diagonal
    v8f s = {};
#pragma unroll
    for (int k0 = 0; k0 < HH; k0 += 32)
      s = wmma_f16(load_a_frag(Q, HH, row0, k0), load_b_frag(K, HH, it * 16, k0), s);
    const int ig = it * 16 + (lane & 15);   // column i
#pragma unroll
    for (int r = 0; r < 8; ++r) {
      int   jg = row0 + r + 8 * (lane >> 4);
      float sv = s[r];
      if (ig >= jg && sv != 0.0f) {         // triu + zero->-inf quirk
        float v = sv * scale;
        if (v > mx[r]) { sm[r] = sm[r] * __expf(mx[r] - v) + 1.0f; mx[r] = v; }
        else             sm[r] += __expf(v - mx[r]);
      }
    }
  }
#pragma unroll
  for (int r = 0; r < 8; ++r) {             // reduce across 16-lane half-wave
#pragma unroll
    for (int off = 1; off < 16; off <<= 1) {
      float om = __shfl_xor(mx[r], off, 32);
      float os = __shfl_xor(sm[r], off, 32);
      if (om > mx[r]) { sm[r] = sm[r] * __expf(mx[r] - om) + os; mx[r] = om; }
      else if (om != -INFINITY) sm[r] += os * __expf(om - mx[r]);
    }
  }
  if ((lane & 15) == 0) {
#pragma unroll
    for (int r = 0; r < 8; ++r) {
      int jg = row0 + r + 8 * (lane >> 4);
      mbuf[(size_t)b * NN + jg] = mx[r];
      zbuf[(size_t)b * NN + jg] = sm[r];
    }
  }
}

// ---- Pass B: out[i,h] = sum_{j<=i} exp(s[j,i]*scale - m[j])/Z[j] * v[j,h] ---
__global__ __launch_bounds__(32) void out_kernel(
    const _Float16* __restrict__ kh, const _Float16* __restrict__ qh,
    const _Float16* __restrict__ vt, const float* __restrict__ mbuf,
    const float* __restrict__ zbuf, float* __restrict__ out) {
  const int it    = blockIdx.x % (NN / 16);
  const int b     = blockIdx.x / (NN / 16);
  const int lane  = threadIdx.x & 31;
  const int irow0 = it * 16;
  const float scale = 0.125f;
  __shared__ __align__(16) _Float16 Pt[16 * 32];     // P^T tile relayout buffer
  const _Float16* K = kh + (size_t)b * NN * HH;
  const _Float16* Q = qh + (size_t)b * NN * HH;
  const _Float16* V = vt + (size_t)b * HH * NN;      // [H][N], col-K-contiguous
  v8f a0 = {}, a1 = {}, a2 = {}, a3 = {};
  const int jbmax = (irow0 + 15) >> 5;               // j-blocks of 32, j <= i
  for (int jb = 0; jb <= jbmax; ++jb) {
#pragma unroll
    for (int sub = 0; sub < 2; ++sub) {
      // S'[i_local, j_local] = k[i] . q[j]
      v8f s = {};
#pragma unroll
      for (int k0 = 0; k0 < HH; k0 += 32)
        s = wmma_f16(load_a_frag(K, HH, irow0, k0),
                     load_b_frag(Q, HH, jb * 32 + sub * 16, k0), s);
      int   jg = jb * 32 + sub * 16 + (lane & 15);   // column j (per lane)
      float mj = mbuf[(size_t)b * NN + jg];
      float zj = zbuf[(size_t)b * NN + jg];
      float iz = (zj != 0.f) ? 1.f / zj : 0.f;
      bool  jv = (mj != -INFINITY);
#pragma unroll
      for (int r = 0; r < 8; ++r) {
        int   il = r + 8 * (lane >> 4);
        int   ig = irow0 + il;
        float sv = s[r];
        float p  = 0.f;
        if (jv && ig >= jg && sv != 0.f) p = __expf(sv * scale - mj) * iz;
        Pt[il * 32 + sub * 16 + (lane & 15)] = (_Float16)p;
      }
    }
    __syncthreads();
    v16h pa = load_a_frag(Pt, 32, 0, 0);             // re-layout P^T as A-frag
    v16h b0 = load_b_frag(V, NN,  0, jb * 32);
    v16h b1 = load_b_frag(V, NN, 16, jb * 32);
    v16h b2 = load_b_frag(V, NN, 32, jb * 32);
    v16h b3 = load_b_frag(V, NN, 48, jb * 32);
    a0 = wmma_f16(pa, b0, a0);
    a1 = wmma_f16(pa, b1, a1);
    a2 = wmma_f16(pa, b2, a2);
    a3 = wmma_f16(pa, b3, a3);
    __syncthreads();
  }
#pragma unroll
  for (int r = 0; r < 8; ++r) {
    int row = irow0 + r + 8 * (lane >> 4);
    int h   = lane & 15;
    size_t o = ((size_t)b * NN + row) * HH;
    out[o +      h] = a0[r];
    out[o + 16 + h] = a1[r];
    out[o + 32 + h] = a2[r];
    out[o + 48 + h] = a3[r];
  }
}

// ---- launcher ---------------------------------------------------------------
extern "C" void kernel_launch(void* const* d_in, const int* in_sizes, int n_in,
                              void* d_out, int out_size, void* d_ws, size_t ws_size,
                              hipStream_t stream) {
  const float* x  = (const float*)d_in[0];
  const float* Wq = (const float*)d_in[1];
  const float* bq = (const float*)d_in[2];
  const float* Wk = (const float*)d_in[3];
  const float* bk = (const float*)d_in[4];
  const float* Wv = (const float*)d_in[5];
  const float* bv = (const float*)d_in[6];
  float* out = (float*)d_out;

  char* w = (char*)d_ws;                                   // 20,381,696 B total
  _Float16* xh = (_Float16*)w; w += (size_t)BB * NN * DD * 2;   // 16 MB
  _Float16* Wt = (_Float16*)w; w += (size_t)3 * HH * DD * 2;    // 384 KB
  _Float16* qh = (_Float16*)w; w += (size_t)BB * NN * HH * 2;   // 1 MB
  _Float16* kh = (_Float16*)w; w += (size_t)BB * NN * HH * 2;   // 1 MB
  _Float16* vt = (_Float16*)w; w += (size_t)BB * NN * HH * 2;   // 1 MB (v^T)
  float* mbuf = (float*)w;     w += (size_t)BB * NN * 4;        // 32 KB
  float* zbuf = (float*)w;                                      // 32 KB

  const size_t nx = (size_t)BB * NN * DD;
  cvt_x_kernel<<<dim3((unsigned)(nx / 8 / 256)), dim3(256), 0, stream>>>(x, xh, nx);
  wt_kernel<<<dim3((3 * HH * DD) / 256), dim3(256), 0, stream>>>(Wq, Wk, Wv, Wt);
  qkv_kernel<<<dim3(BB * NN / 16, 3), dim3(128), 0, stream>>>(xh, Wt, bq, bk, bv,
                                                              qh, kh, vt);
  stats_kernel<<<dim3(BB * NN / 16), dim3(32), 0, stream>>>(qh, kh, mbuf, zbuf);
  out_kernel<<<dim3(BB * NN / 16), dim3(32), 0, stream>>>(kh, qh, vt, mbuf, zbuf, out);
}
